// ESN_18640158065251
// MI455X (gfx1250) — compile-verified, matrix operand order checked by reading, not run on hardware
//
#include <hip/hip_runtime.h>
#include <hip/hip_bf16.h>
#include <math.h>

// ---------------- problem constants (from reference) ----------------
#define TDIM 8192
#define DIN  128
#define HDIM 2048
#define DOUT 128

#define NBLK_REC 128
#define ROWS_PER_BLK (HDIM / NBLK_REC)   // 16 rows per block
#define ROWS_PER_WAVE (ROWS_PER_BLK / 8) // 2 rows per wave (8 waves/block)

typedef float v2f  __attribute__((ext_vector_type(2)));
typedef float v8f  __attribute__((ext_vector_type(8)));
typedef __bf16 bf16x16 __attribute__((ext_vector_type(16)));
typedef int v4i __attribute__((vector_size(4 * sizeof(int))));

#if __has_builtin(__builtin_amdgcn_wmma_f32_16x16x4_f32)
#define HAVE_WMMA_F32X4 1
#else
#define HAVE_WMMA_F32X4 0
#endif

#if __has_builtin(__builtin_amdgcn_global_load_async_to_lds_b128) && \
    __has_builtin(__builtin_amdgcn_s_wait_asynccnt)
#define HAVE_ASYNC_LDS 1
#else
#define HAVE_ASYNC_LDS 0
#endif

#if HAVE_WMMA_F32X4
#define KB 16
#else
#define KB 32
#endif
#define LDSROW 132   // 128 + 4 pad vs bank conflicts (132*4 B = multiple of 16)

// builtin signature (from hipcc diagnostic): (v4i AS1*, v4i AS3*, imm, imm)
#define AS1(p) ((__attribute__((address_space(1))) v4i*)(uintptr_t)(p))
#define AS3(p) ((__attribute__((address_space(3))) v4i*)(uintptr_t)(p))

// ---------------- init: zero barrier + activation ping-pong ----------------
__global__ void esn_init_kernel(unsigned* __restrict__ bar,
                                float* __restrict__ actbuf) {
  int i = blockIdx.x * blockDim.x + threadIdx.x;
  if (i < 2) bar[i] = 0u;
  for (int k = i; k < 2 * HDIM; k += gridDim.x * blockDim.x) actbuf[k] = 0.0f;
}

// ---------------- device-scope split grid barrier ----------------
__device__ inline void grid_sync(unsigned* bar, unsigned nblk) {
  __builtin_amdgcn_fence(__ATOMIC_RELEASE, "agent");  // release own stores
  __syncthreads();
  if (threadIdx.x == 0) {
    unsigned gen = __hip_atomic_load(&bar[1], __ATOMIC_RELAXED,
                                     __HIP_MEMORY_SCOPE_AGENT);
    unsigned arrived = __hip_atomic_fetch_add(&bar[0], 1u, __ATOMIC_ACQ_REL,
                                              __HIP_MEMORY_SCOPE_AGENT);
    if (arrived == nblk - 1u) {
      __hip_atomic_store(&bar[0], 0u, __ATOMIC_RELAXED,
                         __HIP_MEMORY_SCOPE_AGENT);
      __hip_atomic_fetch_add(&bar[1], 1u, __ATOMIC_RELEASE,
                             __HIP_MEMORY_SCOPE_AGENT);
    } else {
      while (__hip_atomic_load(&bar[1], __ATOMIC_RELAXED,
                               __HIP_MEMORY_SCOPE_AGENT) == gen) {
        __builtin_amdgcn_s_sleep(2);
      }
    }
  }
  __syncthreads();
  __builtin_amdgcn_fence(__ATOMIC_ACQUIRE, "agent");  // invalidate before reads
}

// ---------------- persistent recurrence kernel ----------------
// hid_t = (inp_t . W_ih^T + b) + act_{t-1} . W_hh^T ; act_t = tanh(hid_t)
// W_hh (16.8 MB fp32) stays resident in 192 MB L2; act double-buffered global.
__global__ __launch_bounds__(256) void esn_recurrence_kernel(
    const float* __restrict__ inp,    // T x DIN
    const float* __restrict__ w_ih,   // H x DIN
    const float* __restrict__ b_ih,   // H
    const float* __restrict__ w_hh,   // H x H
    float* __restrict__ hid,          // T x H   (workspace)
    float* __restrict__ actbuf,       // 2 x H   (workspace, zeroed)
    unsigned* __restrict__ bar) {
  __shared__ __align__(16) float s_act[HDIM];               // 8 KB
  __shared__ __align__(16) float s_inp[DIN];                // 512 B
  __shared__ __align__(16) float s_wih[ROWS_PER_BLK][DIN];  // 8 KB
  __shared__ float s_b[ROWS_PER_BLK];

  const int tid  = threadIdx.x;
  const int lane = tid & 31;
  const int wave = tid >> 5;                       // 0..7
  const int row0 = blockIdx.x * ROWS_PER_BLK;

  // stage this block's W_ih rows + bias once (b128 loads)
  for (int i = tid; i < ROWS_PER_BLK * DIN / 4; i += 256)
    ((float4*)s_wih)[i] = ((const float4*)(w_ih + (size_t)row0 * DIN))[i];
  if (tid < ROWS_PER_BLK) s_b[tid] = b_ih[row0 + tid];

  for (int t = 0; t < TDIM; ++t) {
    const float* act_cur = actbuf + (size_t)(t & 1) * HDIM;
    float*       act_nxt = actbuf + (size_t)((t + 1) & 1) * HDIM;

    __syncthreads();  // protect LDS reuse across iterations
    for (int i = tid; i < HDIM / 4; i += 256)
      ((float4*)s_act)[i] = ((const float4*)act_cur)[i];
    if (tid < DIN / 4)
      ((float4*)s_inp)[tid] = ((const float4*)(inp + (size_t)t * DIN))[tid];
    __syncthreads();

    for (int rr = 0; rr < ROWS_PER_WAVE; ++rr) {
      const int r   = wave * ROWS_PER_WAVE + rr;   // 0..15
      const int row = row0 + r;
      const float4* w4 = (const float4*)(w_hh + (size_t)row * HDIM);
      const float4* a4 = (const float4*)s_act;
      float p = 0.0f;
      // 2048-long dot: coalesced b128 per lane, L2-hot (16 iterations)
      for (int i = lane; i < HDIM / 4; i += 32) {
        float4 w = w4[i];
        float4 a = a4[i];
        p = fmaf(a.x, w.x, p);
        p = fmaf(a.y, w.y, p);
        p = fmaf(a.z, w.z, p);
        p = fmaf(a.w, w.w, p);
      }
      {  // fused input projection: DIN/4 == 32 lanes -> single b128 each
        float4 wi = ((const float4*)s_wih[r])[lane];
        float4 ui = ((const float4*)s_inp)[lane];
        p = fmaf(ui.x, wi.x, p);
        p = fmaf(ui.y, wi.y, p);
        p = fmaf(ui.z, wi.z, p);
        p = fmaf(ui.w, wi.w, p);
      }
      // wave32 butterfly reduction
      for (int off = 16; off > 0; off >>= 1) p += __shfl_xor(p, off, 32);
      if (lane == 0) {
        float nh = p + s_b[r];
        hid[(size_t)t * HDIM + row] = nh;
        act_nxt[row] = tanhf(nh);
      }
    }
    grid_sync(bar, NBLK_REC);
  }
}

// ---------------- async K-panel staging (global -> LDS, ASYNCcnt) ----------
__device__ __forceinline__ void stage_chunk(
    const float* __restrict__ A, const float* __restrict__ B, int ldb,
    int wash, int K, int i0, int j0, int k0,
    float (*__restrict__ Asb)[LDSROW], float (*__restrict__ Bsb)[LDSROW],
    int tid) {
  for (int c = tid; c < KB * 32; c += 256) {     // 16-byte chunks
    const int kk  = c >> 5;
    const int col = (c & 31) << 2;
    const int t   = k0 + kk;
    float* lA = &Asb[kk][col];
    float* lB = &Bsb[kk][col];
    if (t < K) {
      const float* gA = A + (size_t)(wash + t) * HDIM + i0 + col;
      const float* gB = B + (size_t)(wash + t) * ldb  + j0 + col;
#if HAVE_ASYNC_LDS
      __builtin_amdgcn_global_load_async_to_lds_b128(AS1(gA), AS3(lA), 0, 0);
      __builtin_amdgcn_global_load_async_to_lds_b128(AS1(gB), AS3(lB), 0, 0);
#else
      *(float4*)lA = *(const float4*)gA;
      *(float4*)lB = *(const float4*)gB;
#endif
    } else {  // zero-pad K tail (contributes 0 to the accumulation)
      *(float4*)lA = make_float4(0.f, 0.f, 0.f, 0.f);
      *(float4*)lB = make_float4(0.f, 0.f, 0.f, 0.f);
    }
  }
}

// ---------------- blocked A^T*B GEMM over time axis (WMMA) ----------------
// C[i,j] = sum_{t=washout..T-1} A[t,i] * B[t,j]
// 128x128 macro-tile / workgroup; 8 waves, each a 2x4 grid of 16x16 tiles.
// Double-buffered LDS panels filled by async global->LDS copies.
__global__ __launch_bounds__(256) void esn_gram_gemm_kernel(
    const float* __restrict__ A,      // T x HDIM (hid)
    const float* __restrict__ B,      // T x ldb  (hid or target)
    const int* __restrict__ washout_p,
    int ldb, float* __restrict__ C, int ldc) {
  const int wash = *washout_p;
  const int K = TDIM - wash;
  const int i0 = blockIdx.y * 128;
  const int j0 = blockIdx.x * 128;

  __shared__ __align__(16) float As[2][KB][LDSROW];
  __shared__ __align__(16) float Bs[2][KB][LDSROW];

  const int tid   = threadIdx.x;
  const int lane  = tid & 31;
  const int wave  = tid >> 5;
  const int wrow  = wave & 3;     // tile rows 2*wrow, 2*wrow+1
  const int wcol  = wave >> 2;    // tile cols 4*wcol .. 4*wcol+3
  const int li    = lane & 15;
  const int lhalf = (lane < 16) ? 0 : 1;

  v8f acc[2][4];
  for (int r = 0; r < 2; ++r)
    for (int c = 0; c < 4; ++c) acc[r][c] = (v8f)(0.0f);

  const int nchunks = (K + KB - 1) / KB;
  stage_chunk(A, B, ldb, wash, K, i0, j0, 0, As[0], Bs[0], tid);

  for (int kc = 0; kc < nchunks; ++kc) {
    const int cur = kc & 1;
#if HAVE_ASYNC_LDS
    __builtin_amdgcn_s_wait_asynccnt(0);   // chunk kc landed in LDS
#endif
    __syncthreads();  // panel visible; also proves compute(kc-1) is done
    if (kc + 1 < nchunks)
      stage_chunk(A, B, ldb, wash, K, i0, j0, (kc + 1) * KB,
                  As[1 - cur], Bs[1 - cur], tid);

#if HAVE_WMMA_F32X4
    for (int kk = 0; kk < KB; kk += 4) {
      v2f a[2], b[4];
      for (int r = 0; r < 2; ++r) {
        int irow = (2 * wrow + r) * 16 + li;
        a[r].x = As[cur][kk + 2 * lhalf + 0][irow];   // K = 0 / 2
        a[r].y = As[cur][kk + 2 * lhalf + 1][irow];   // K = 1 / 3
      }
      for (int c = 0; c < 4; ++c) {
        int jcol = (4 * wcol + c) * 16 + li;
        b[c].x = Bs[cur][kk + lhalf + 0][jcol];       // K = 0 / 1
        b[c].y = Bs[cur][kk + lhalf + 2][jcol];       // K = 2 / 3
      }
      for (int r = 0; r < 2; ++r)
        for (int c = 0; c < 4; ++c)
          acc[r][c] = __builtin_amdgcn_wmma_f32_16x16x4_f32(
              false, a[r], false, b[c], (short)0, acc[r][c], false, false);
    }
#else
    {
      bf16x16 a[2], b[4];
      for (int r = 0; r < 2; ++r) {
        int irow = (2 * wrow + r) * 16 + li;
        for (int e = 0; e < 16; ++e) {
          int v = e >> 1, p = e & 1;
          int kbase = (v < 4) ? (2 * v) : (16 + 2 * (v - 4));
          a[r][e] = (__bf16)As[cur][kbase + p + 8 * lhalf][irow];
        }
      }
      for (int c = 0; c < 4; ++c) {
        int jcol = (4 * wcol + c) * 16 + li;
        for (int e = 0; e < 16; ++e) {
          int v = e >> 1, p = e & 1;
          int kbase = (v < 4) ? (2 * v) : (16 + 2 * (v - 4));
          b[c][e] = (__bf16)Bs[cur][kbase + p + 8 * lhalf][jcol];
        }
      }
      for (int r = 0; r < 2; ++r)
        for (int c = 0; c < 4; ++c)
          acc[r][c] = __builtin_amdgcn_wmma_f32_16x16x32_bf16(
              false, a[r], false, b[c], (short)0, acc[r][c], false, false);
    }
#endif
  }

  // C/D layout: VGPR v -> M=v (lanes 0-15) / M=v+8 (lanes 16-31), N=lane%16
  const int mofs = (lane < 16) ? 0 : 8;
  for (int r = 0; r < 2; ++r) {
    for (int c = 0; c < 4; ++c) {
      int itile = i0 + (2 * wrow + r) * 16;
      int jtile = j0 + (4 * wcol + c) * 16;
      for (int v = 0; v < 8; ++v)
        C[(size_t)(itile + mofs + v) * ldc + jtile + li] = acc[r][c][v];
    }
  }
}

// ---------------- host launch ----------------
extern "C" void kernel_launch(void* const* d_in, const int* in_sizes, int n_in,
                              void* d_out, int out_size, void* d_ws,
                              size_t ws_size, hipStream_t stream) {
  (void)in_sizes; (void)n_in; (void)out_size; (void)ws_size;
  const float* inp     = (const float*)d_in[0];  // 8192 x 128
  const float* target  = (const float*)d_in[1];  // 8192 x 128
  const float* w_ih    = (const float*)d_in[2];  // 2048 x 128
  const float* b_ih    = (const float*)d_in[3];  // 2048
  const float* w_hh    = (const float*)d_in[4];  // 2048 x 2048
  const int*   washout = (const int*)d_in[5];    // scalar on device

  char* ws = (char*)d_ws;
  unsigned* bar   = (unsigned*)ws;                                   // 256 B
  float* actbuf   = (float*)(ws + 256);                              // 2 x H
  float* hid      = (float*)(ws + 256 + 2 * HDIM * sizeof(float));   // T x H

  float* HTH = (float*)d_out;                       // 2048 x 2048
  float* HTY = HTH + (size_t)HDIM * HDIM;           // 2048 x 128

  esn_init_kernel<<<16, 256, 0, stream>>>(bar, actbuf);
  esn_recurrence_kernel<<<NBLK_REC, 256, 0, stream>>>(inp, w_ih, b_ih, w_hh,
                                                      hid, actbuf, bar);
  dim3 gHTH(HDIM / 128, HDIM / 128);   // 16 x 16
  esn_gram_gemm_kernel<<<gHTH, 256, 0, stream>>>(hid, hid, washout, HDIM, HTH,
                                                 HDIM);
  dim3 gHTY(DOUT / 128, HDIM / 128);   // 1 x 16
  esn_gram_gemm_kernel<<<gHTY, 256, 0, stream>>>(hid, target, washout, DOUT,
                                                 HTY, DOUT);
}
